// InteractionBlock_3367254360714
// MI455X (gfx1250) — matrix operand order repaired and checked by reading, not compiled, not run on hardware
//
#include <hip/hip_runtime.h>

// ---------------------------------------------------------------------------
// DimeNet InteractionBlock for MI455X (gfx1250, wave32, WMMA).
// All GEMMs use v_wmma_f32_16x16x32_bf16 with fp32 accumulation.
// ---------------------------------------------------------------------------

typedef __attribute__((ext_vector_type(16))) __bf16 bf16x16;
typedef __attribute__((ext_vector_type(8)))  float  f32x8;

union Frag {
    bf16x16 v;
    uint4   q[2];
    __bf16  e[16];
};

__device__ __forceinline__ float silu_f(float v) {
    return v * (1.0f / (1.0f + __expf(-v)));
}

__device__ __forceinline__ void tile_zero(f32x8 t[8]) {
#pragma unroll
    for (int i = 0; i < 8; ++i)
#pragma unroll
        for (int j = 0; j < 8; ++j) t[i][j] = 0.0f;
}

__device__ __forceinline__ void tile_add(f32x8 a[8], const f32x8 b[8]) {
#pragma unroll
    for (int i = 0; i < 8; ++i) a[i] += b[i];
}

// Load a 16x128 fp32 tile from global in WMMA C-layout (row = r + 8*half).
__device__ __forceinline__ void tile_load(f32x8 t[8], const float* __restrict__ base,
                                          int m0, int nrows, int lane) {
    const int n0 = lane & 15, hf = lane >> 4;
#pragma unroll
    for (int nt = 0; nt < 8; ++nt)
#pragma unroll
        for (int r = 0; r < 8; ++r) {
            int row   = m0 + r + hf * 8;
            t[nt][r] = (row < nrows) ? base[(size_t)row * 128 + nt * 16 + n0] : 0.0f;
        }
}

__device__ __forceinline__ void tile_store(const f32x8 t[8], float* __restrict__ base,
                                           int m0, int nrows, int lane) {
    const int n0 = lane & 15, hf = lane >> 4;
#pragma unroll
    for (int nt = 0; nt < 8; ++nt)
#pragma unroll
        for (int r = 0; r < 8; ++r) {
            int row = m0 + r + hf * 8;
            if (row < nrows) base[(size_t)row * 128 + nt * 16 + n0] = t[nt][r];
        }
}

// Dump a C-layout tile into a 16x128 bf16 LDS slab (row-major) and fence DS.
__device__ __forceinline__ void tile_to_lds(__bf16* slab, const f32x8 t[8], int lane) {
    const int n0 = lane & 15, hf = lane >> 4;
#pragma unroll
    for (int nt = 0; nt < 8; ++nt)
#pragma unroll
        for (int r = 0; r < 8; ++r)
            slab[(r + hf * 8) * 128 + nt * 16 + n0] = (__bf16)t[nt][r];
    asm volatile("s_wait_dscnt 0" ::: "memory");
}

// out = silu(acc + bias), optionally silu twice (the lin_ji / lin_kj path).
__device__ __forceinline__ void bias_silu(f32x8 t[8], const float* __restrict__ bias,
                                          int lane, int twice) {
    const int n0 = lane & 15;
#pragma unroll
    for (int nt = 0; nt < 8; ++nt) {
        float bv = bias[nt * 16 + n0];
#pragma unroll
        for (int r = 0; r < 8; ++r) {
            float v = silu_f(t[nt][r] + bv);
            if (twice) v = silu_f(v);
            t[nt][r] = v;
        }
    }
}

// acc += A(16x128, bf16 LDS slab) @ W(128x128), W stored bf16 [N][K] (K-contiguous).
__device__ __forceinline__ void gemm16x128(f32x8 acc[8], const __bf16* slab,
                                           const __bf16* __restrict__ Wt, int lane) {
    const int r  = lane & 15;
    const int hf = lane >> 4;
#pragma unroll
    for (int kc = 0; kc < 4; ++kc) {
        Frag a;
        const __bf16* pa = slab + r * 128 + kc * 32 + hf * 8;   // K runs: [0..7], [16..23] (+8 on upper half)
        a.q[0] = *(const uint4*)(pa);
        a.q[1] = *(const uint4*)(pa + 16);
#pragma unroll
        for (int nt = 0; nt < 8; ++nt) {
            Frag b;
            const __bf16* pb = Wt + (size_t)(r + nt * 16) * 128 + kc * 32 + hf * 8;
            b.q[0] = *(const uint4*)(pb);
            b.q[1] = *(const uint4*)(pb + 16);
            acc[nt] = __builtin_amdgcn_wmma_f32_16x16x32_bf16(
                false, a.v, false, b.v, (short)0, acc[nt], false, false);
        }
    }
}

// ---------------------------------------------------------------------------
// Prep kernels
// ---------------------------------------------------------------------------
__global__ void k_wt(const float* __restrict__ W, __bf16* __restrict__ Wt) {
    int g = blockIdx.x * 256 + threadIdx.x;          // 128*128 elements
    if (g >= 128 * 128) return;
    int n = g >> 7, k = g & 127;
    Wt[g] = (__bf16)W[k * 128 + n];                  // transpose to [N][K]
}

__global__ void k_cvt(const float* __restrict__ src, __bf16* __restrict__ dst, int n) {
    int g = blockIdx.x * 256 + threadIdx.x;
    if (g < n) dst[g] = (__bf16)src[g];
}

__global__ void k_rbfh(const float* __restrict__ rbf, const float* __restrict__ rbfW,
                       float* __restrict__ rbfh, int E) {
    int g = blockIdx.x * 256 + threadIdx.x;
    if (g >= E * 128) return;
    int e = g >> 7, n = g & 127;
    float s = 0.0f;
#pragma unroll
    for (int q = 0; q < 16; ++q) s += rbf[e * 16 + q] * rbfW[q * 128 + n];
    rbfh[g] = s;
}

__global__ void k_sbfh(const float* __restrict__ sbf, const float* __restrict__ sbfW,
                       float* __restrict__ sbfh, int T) {
    int g = blockIdx.x * 256 + threadIdx.x;
    if (g >= T * 8) return;
    int t = g >> 3, o = g & 7;
    float s = 0.0f;
    for (int q = 0; q < 112; ++q) s += sbf[(size_t)t * 112 + q] * sbfW[q * 8 + o];
    sbfh[g] = s;
}

// ---------------------------------------------------------------------------
// Edge transforms: x_ji -> hbuf (acts as agg init), x_kj*rbf_h -> bf16
// ---------------------------------------------------------------------------
__global__ void __launch_bounds__(256)
k_edge(const float* __restrict__ x, const float* __restrict__ rbfh,
       const __bf16* __restrict__ WtJi, const __bf16* __restrict__ WtKj,
       const float* __restrict__ jiB, const float* __restrict__ kjB,
       float* __restrict__ hbuf, __bf16* __restrict__ xkjbf, int E) {
    __shared__ __bf16 xt[128 * 128];
    const int tid = threadIdx.x;
    const int e0  = blockIdx.x * 128;
#pragma unroll
    for (int i = 0; i < 64; ++i) {
        int idx = i * 256 + tid;
        int row = e0 + (idx >> 7);
        xt[idx] = (row < E) ? (__bf16)x[(size_t)row * 128 + (idx & 127)] : (__bf16)0.0f;
    }
    __syncthreads();

    const int wave = tid >> 5, lane = tid & 31;
    const __bf16* slab = xt + wave * 16 * 128;
    const int m0 = e0 + wave * 16;
    const int n0 = lane & 15, hf = lane >> 4;

    f32x8 acc[8];
    tile_zero(acc);
    gemm16x128(acc, slab, WtJi, lane);
    bias_silu(acc, jiB, lane, 1);                     // double silu
#pragma unroll
    for (int nt = 0; nt < 8; ++nt)
#pragma unroll
        for (int r = 0; r < 8; ++r) {
            int row = m0 + r + hf * 8;
            if (row < E) hbuf[(size_t)row * 128 + nt * 16 + n0] = acc[nt][r];
        }

    tile_zero(acc);
    gemm16x128(acc, slab, WtKj, lane);
    bias_silu(acc, kjB, lane, 1);
#pragma unroll
    for (int nt = 0; nt < 8; ++nt)
#pragma unroll
        for (int r = 0; r < 8; ++r) {
            int row = m0 + r + hf * 8;
            if (row < E) {
                size_t off = (size_t)row * 128 + nt * 16 + n0;
                xkjbf[off] = (__bf16)(acc[nt][r] * rbfh[off]);
            }
        }
}

// ---------------------------------------------------------------------------
// Triplets: m[t,o] = sum_b sbf_h[t,b] * (x_kj[kj] @ bil_W[:,b,:]^T)[t,o]
// A-fragments (x_kj rows) are b-invariant: loaded once, 8 K=128 GEMMs,
// fp32 scale-accumulate of each partial result. Scatter-add via idx_ji.
// ---------------------------------------------------------------------------
__global__ void __launch_bounds__(256)
k_tri(const __bf16* __restrict__ xkjbf, const float* __restrict__ sbfh,
      const int* __restrict__ idx_kj, const int* __restrict__ idx_ji,
      const __bf16* __restrict__ bilbf, float* __restrict__ hbuf, int T) {
    __shared__ __bf16 xkt[128 * 128];
    __shared__ float  sbt[128 * 8];
    const int tid = threadIdx.x;
    const int t0  = blockIdx.x * 128;
    {   // gather x_kj rows (bf16, 256B each); 2 threads per row
        int row = tid >> 1, half = tid & 1;
        int t   = t0 + row;
        int src = (t < T) ? idx_kj[t] : 0;
        const uint4* s = (const uint4*)(xkjbf + (size_t)src * 128 + half * 64);
        uint4* d = (uint4*)(xkt + row * 128 + half * 64);
#pragma unroll
        for (int i = 0; i < 8; ++i) d[i] = s[i];
    }
    {   // sbf_h tile
        int t = t0 + (tid >> 1);
        float4 v = make_float4(0.f, 0.f, 0.f, 0.f);
        if (t < T) v = ((const float4*)(sbfh + (size_t)t0 * 8))[tid];
        ((float4*)sbt)[tid] = v;
    }
    __syncthreads();

    const int wave = tid >> 5, lane = tid & 31;
    const int r = lane & 15, hf = lane >> 4;
    const int trow = wave * 16 + r;

    // A-fragments: x_kj rows for this wave's 16 triplets (b-invariant).
    Frag afr[4];
#pragma unroll
    for (int kc = 0; kc < 4; ++kc) {
        const __bf16* pa = xkt + trow * 128 + kc * 32 + hf * 8;
        afr[kc].q[0] = *(const uint4*)(pa);
        afr[kc].q[1] = *(const uint4*)(pa + 16);
    }

    f32x8 acc[8];
    tile_zero(acc);
    for (int b = 0; b < 8; ++b) {
        f32x8 y[8];
        tile_zero(y);
#pragma unroll
        for (int kc = 0; kc < 4; ++kc) {
#pragma unroll
            for (int nt = 0; nt < 8; ++nt) {
                Frag bw;
                const __bf16* pb = bilbf + (size_t)(r + nt * 16) * 1024 +
                                   b * 128 + kc * 32 + hf * 8;
                bw.q[0] = *(const uint4*)(pb);
                bw.q[1] = *(const uint4*)(pb + 16);
                y[nt] = __builtin_amdgcn_wmma_f32_16x16x32_bf16(
                    false, afr[kc].v, false, bw.v, (short)0, y[nt], false, false);
            }
        }
#pragma unroll
        for (int rr = 0; rr < 8; ++rr) {
            float s = sbt[(wave * 16 + rr + hf * 8) * 8 + b];
#pragma unroll
            for (int nt = 0; nt < 8; ++nt)
                acc[nt][rr] += s * y[nt][rr];
        }
    }

    int eji[8];
#pragma unroll
    for (int rr = 0; rr < 8; ++rr) {
        int t = t0 + wave * 16 + rr + hf * 8;
        eji[rr] = (t < T) ? idx_ji[t] : -1;
    }
    const int n0 = lane & 15;
#pragma unroll
    for (int nt = 0; nt < 8; ++nt)
#pragma unroll
        for (int rr = 0; rr < 8; ++rr)
            if (eji[rr] >= 0)
                unsafeAtomicAdd(&hbuf[(size_t)eji[rr] * 128 + nt * 16 + n0], acc[nt][rr]);
}

// ---------------------------------------------------------------------------
// Residual chain: 7 GEMMs, wave-private slab, no inter-wave sync needed
// ---------------------------------------------------------------------------
__global__ void __launch_bounds__(256)
k_post(const float* __restrict__ hbuf, const float* __restrict__ x,
       const __bf16* __restrict__ Wp1, const float* __restrict__ b1,
       const __bf16* __restrict__ Wp2, const float* __restrict__ b2,
       const __bf16* __restrict__ Wp3, const float* __restrict__ b3,
       const __bf16* __restrict__ Wq1, const float* __restrict__ c1,
       const __bf16* __restrict__ Wq2, const float* __restrict__ c2,
       const __bf16* __restrict__ Wq3, const float* __restrict__ c3,
       const __bf16* __restrict__ Wq4, const float* __restrict__ c4,
       float* __restrict__ out, int E) {
    __shared__ __bf16 tl[8][16 * 128];
    const int tid = threadIdx.x, wave = tid >> 5, lane = tid & 31;
    __bf16* slab = tl[wave];
    const int m0 = blockIdx.x * 128 + wave * 16;

    f32x8 h[8], acc[8], res[8];
    tile_load(h, hbuf, m0, E, lane);                  // h = x_ji + agg
    tile_to_lds(slab, h, lane);
    tile_zero(acc); gemm16x128(acc, slab, Wp1, lane); bias_silu(acc, b1, lane, 0);
    tile_to_lds(slab, acc, lane);
    tile_zero(acc); gemm16x128(acc, slab, Wp2, lane); bias_silu(acc, b2, lane, 0);
    tile_add(h, acc);                                  // h1 = h + ...
    tile_to_lds(slab, h, lane);
    tile_zero(acc); gemm16x128(acc, slab, Wp3, lane); bias_silu(acc, b3, lane, 0); // h2
    tile_load(res, x, m0, E, lane);
    tile_add(res, acc);                                // out0 = h2 + x
    tile_to_lds(slab, res, lane);
    tile_zero(acc); gemm16x128(acc, slab, Wq1, lane); bias_silu(acc, c1, lane, 0);
    tile_to_lds(slab, acc, lane);
    tile_zero(acc); gemm16x128(acc, slab, Wq2, lane); bias_silu(acc, c2, lane, 0);
    tile_add(res, acc);                                // out1
    tile_to_lds(slab, res, lane);
    tile_zero(acc); gemm16x128(acc, slab, Wq3, lane); bias_silu(acc, c3, lane, 0);
    tile_to_lds(slab, acc, lane);
    tile_zero(acc); gemm16x128(acc, slab, Wq4, lane); bias_silu(acc, c4, lane, 0);
    tile_add(res, acc);                                // out
    tile_store(res, out, m0, E, lane);
}

// ---------------------------------------------------------------------------
extern "C" void kernel_launch(void* const* d_in, const int* in_sizes, int n_in,
                              void* d_out, int out_size, void* d_ws, size_t ws_size,
                              hipStream_t stream) {
    const float* x    = (const float*)d_in[0];
    const float* rbf  = (const float*)d_in[1];
    const float* sbf  = (const float*)d_in[2];
    const int* kj_idx = (const int*)d_in[3];
    const int* ji_idx = (const int*)d_in[4];
    const float* rbfW = (const float*)d_in[5];
    const float* sbfW = (const float*)d_in[6];
    const float* jiW  = (const float*)d_in[7];
    const float* jiB  = (const float*)d_in[8];
    const float* kjW  = (const float*)d_in[9];
    const float* kjB  = (const float*)d_in[10];
    const float* bilW = (const float*)d_in[11];
    const float* pW1 = (const float*)d_in[12]; const float* pb1 = (const float*)d_in[13];
    const float* pW2 = (const float*)d_in[14]; const float* pb2 = (const float*)d_in[15];
    const float* pW3 = (const float*)d_in[16]; const float* pb3 = (const float*)d_in[17];
    const float* qW1 = (const float*)d_in[18]; const float* qb1 = (const float*)d_in[19];
    const float* qW2 = (const float*)d_in[20]; const float* qb2 = (const float*)d_in[21];
    const float* qW3 = (const float*)d_in[22]; const float* qb3 = (const float*)d_in[23];
    const float* qW4 = (const float*)d_in[24]; const float* qb4 = (const float*)d_in[25];
    (void)n_in; (void)out_size; (void)ws_size;

    const int E = in_sizes[0] / 128;
    const int T = in_sizes[3];

    char* ws = (char*)d_ws;
    size_t off = 0;
    auto take = [&](size_t bytes) -> char* {
        char* p = ws + off;
        off = (off + bytes + 255) & ~(size_t)255;
        return p;
    };
    __bf16* WtJi = (__bf16*)take(16384 * 2);
    __bf16* WtKj = (__bf16*)take(16384 * 2);
    __bf16* Wtp1 = (__bf16*)take(16384 * 2);
    __bf16* Wtp2 = (__bf16*)take(16384 * 2);
    __bf16* Wtp3 = (__bf16*)take(16384 * 2);
    __bf16* Wtq1 = (__bf16*)take(16384 * 2);
    __bf16* Wtq2 = (__bf16*)take(16384 * 2);
    __bf16* Wtq3 = (__bf16*)take(16384 * 2);
    __bf16* Wtq4 = (__bf16*)take(16384 * 2);
    __bf16* bilbf = (__bf16*)take((size_t)131072 * 2);
    float*  rbfh  = (float*)take((size_t)E * 128 * 4);
    float*  sbfh  = (float*)take((size_t)T * 8 * 4);
    __bf16* xkjbf = (__bf16*)take((size_t)E * 128 * 2);
    float*  hbuf  = (float*)take((size_t)E * 128 * 4);

    dim3 blk(256);
    k_wt<<<64, blk, 0, stream>>>(jiW, WtJi);
    k_wt<<<64, blk, 0, stream>>>(kjW, WtKj);
    k_wt<<<64, blk, 0, stream>>>(pW1, Wtp1);
    k_wt<<<64, blk, 0, stream>>>(pW2, Wtp2);
    k_wt<<<64, blk, 0, stream>>>(pW3, Wtp3);
    k_wt<<<64, blk, 0, stream>>>(qW1, Wtq1);
    k_wt<<<64, blk, 0, stream>>>(qW2, Wtq2);
    k_wt<<<64, blk, 0, stream>>>(qW3, Wtq3);
    k_wt<<<64, blk, 0, stream>>>(qW4, Wtq4);
    k_cvt<<<512, blk, 0, stream>>>(bilW, bilbf, 131072);
    k_rbfh<<<(E * 128 + 255) / 256, blk, 0, stream>>>(rbf, rbfW, rbfh, E);
    k_sbfh<<<(T * 8 + 255) / 256, blk, 0, stream>>>(sbf, sbfW, sbfh, T);
    k_edge<<<(E + 127) / 128, blk, 0, stream>>>(x, rbfh, WtJi, WtKj, jiB, kjB,
                                                hbuf, xkjbf, E);
    k_tri<<<(T + 127) / 128, blk, 0, stream>>>(xkjbf, sbfh, kj_idx, ji_idx,
                                               bilbf, hbuf, T);
    k_post<<<(E + 127) / 128, blk, 0, stream>>>(hbuf, x,
                                                Wtp1, pb1, Wtp2, pb2, Wtp3, pb3,
                                                Wtq1, qb1, Wtq2, qb2,
                                                Wtq3, qb3, Wtq4, qb4,
                                                (float*)d_out, E);
}
